// GraphIsomorphismNet_56985626083923
// MI455X (gfx1250) — matrix operand order repaired and checked by reading, not compiled, not run on hardware
//
#include <hip/hip_runtime.h>
#include <hip/hip_bf16.h>
#include <cmath>

typedef __attribute__((ext_vector_type(16))) _Float16 v16h;
typedef __attribute__((ext_vector_type(8)))  _Float16 v8h;
typedef __attribute__((ext_vector_type(8)))  float    v8f;

#define H 128
#define WSTRIDE 136   // 128 + 8 halfs pad: stride-68 words -> conflict-spread banks, keeps 16B align

__device__ __forceinline__ float sigm_(float x) { return 1.0f / (1.0f + expf(-x)); }

__device__ __forceinline__ unsigned enc_ord(float f) {
    unsigned b = __float_as_uint(f);
    return (b & 0x80000000u) ? ~b : (b | 0x80000000u);
}
__device__ __forceinline__ float dec_ord(unsigned u) {
    unsigned b = (u & 0x80000000u) ? (u ^ 0x80000000u) : ~u;
    return __uint_as_float(b);
}

// Stage a 128(out) x 128(k) f16 weight panel into LDS, transposed: Wt[n][k].
// T=1: W stored [Nout, K] row-major (used for "@ W.T"), direct copy (coalesced as-is).
// T=0: W stored [K, Nout] row-major (used for "@ W"); swap thread->(n,k) mapping so
//      consecutive lanes read consecutive n at fixed k (coalesced global reads).
__device__ __forceinline__ void stage_w(_Float16* wl, const float* __restrict__ W,
                                        int T, int ldw, int nbase, int kbase) {
    if (T) {
        for (int i = threadIdx.x; i < H * H; i += 256) {
            int n = i >> 7, k = i & (H - 1);
            wl[n * WSTRIDE + k] = (_Float16)W[(size_t)(nbase + n) * ldw + (size_t)(kbase + k)];
        }
    } else {
        for (int i = threadIdx.x; i < H * H; i += 256) {
            int k = i >> 7, n = i & (H - 1);
            wl[n * WSTRIDE + k] = (_Float16)W[(size_t)(kbase + k) * ldw + (size_t)(nbase + n)];
        }
    }
}

// Accumulate 16 rows x 128 cols over a staged K=128 panel using v_wmma_f32_16x16x32_f16.
// All 8 B-fragments of a k-chunk are hoisted into registers first so the ds_load_b128s
// clause together behind a single s_wait_dscnt, then the 8 WMMAs issue back-to-back.
__device__ __forceinline__ void mma_k128(const float* __restrict__ Arow,
                                         const _Float16* wl, v8f acc[8]) {
    int lane = threadIdx.x & 31;
    int kbo  = (lane & 16) ? 8 : 0;
#pragma unroll
    for (int kc = 0; kc < 4; ++kc) {
        int k0 = kc * 32;
        const float* p = Arow + k0 + kbo;
        v16h a;
#pragma unroll
        for (int e = 0; e < 8; ++e) { a[e] = (_Float16)p[e]; a[8 + e] = (_Float16)p[16 + e]; }
        int kk = k0 + ((lane & 16) ? 8 : 0);
        v16h b[8];
#pragma unroll
        for (int t = 0; t < 8; ++t) {
            const _Float16* bp = wl + (t * 16 + (lane & 15)) * WSTRIDE + kk;
            v8h lo = *(const v8h*)bp;
            v8h hi = *(const v8h*)(bp + 16);
            b[t] = __builtin_shufflevector(lo, hi, 0,1,2,3,4,5,6,7,8,9,10,11,12,13,14,15);
        }
#pragma unroll
        for (int t = 0; t < 8; ++t) {
            acc[t] = __builtin_amdgcn_wmma_f32_16x16x32_f16(
                false, a, false, b[t], (short)0, acc[t], false, false);
        }
    }
}

// OUT[rows,128] = act(A[rows,K] @ Weff + bias),  Weff per T flag. act: 0=none, 1=relu.
__global__ __launch_bounds__(256) void k_wmma_gemm(
    const float* __restrict__ A, int rows, int K,
    const float* __restrict__ W, int T, int ldw,
    const float* __restrict__ bias, float* __restrict__ OUT, int act) {
    __shared__ _Float16 wl[H * WSTRIDE];
    int wave = threadIdx.x >> 5, lane = threadIdx.x & 31;
    int r0 = blockIdx.x * 128 + wave * 16;
    int rowA = r0 + (lane & 15); if (rowA > rows - 1) rowA = rows - 1;
    const float* Abase = A + (size_t)rowA * K;
    v8f acc[8];
#pragma unroll
    for (int t = 0; t < 8; ++t)
#pragma unroll
        for (int j = 0; j < 8; ++j) acc[t][j] = 0.0f;

    for (int kb = 0; kb < K; kb += 128) {
        stage_w(wl, W, T, ldw, 0, kb);
        __syncthreads();
        mma_k128(Abase + kb, wl, acc);
        __syncthreads();
    }
    int rsel = (lane & 16) ? 8 : 0;
#pragma unroll
    for (int t = 0; t < 8; ++t) {
        int col = t * 16 + (lane & 15);
        float bv = bias[col];
#pragma unroll
        for (int rv = 0; rv < 8; ++rv) {
            int row = r0 + rv + rsel;
            if (row < rows) {
                float v = acc[t][rv] + bv;
                if (act) v = fmaxf(v, 0.0f);
                OUT[(size_t)row * H + col] = v;
            }
        }
    }
}

// Fused GRU cell: node (= h) updated in place from m. wih/whh: [3H,H] (@W.T layout).
__global__ __launch_bounds__(256) void k_gru(
    float* __restrict__ node, const float* __restrict__ m,
    const float* __restrict__ wih, const float* __restrict__ whh,
    const float* __restrict__ bih, const float* __restrict__ bhh, int rows) {
    __shared__ _Float16 wl[H * WSTRIDE];
    int wave = threadIdx.x >> 5, lane = threadIdx.x & 31;
    int r0 = blockIdx.x * 128 + wave * 16;
    int rowA = r0 + (lane & 15); if (rowA > rows - 1) rowA = rows - 1;
    const float* Am = m    + (size_t)rowA * H;
    const float* Ah = node + (size_t)rowA * H;

    v8f grz[2][8], gni[8], gnh[8];
#pragma unroll
    for (int g = 0; g < 3; ++g) {
        v8f ai[8], ah[8];
#pragma unroll
        for (int t = 0; t < 8; ++t)
#pragma unroll
            for (int j = 0; j < 8; ++j) { ai[t][j] = 0.0f; ah[t][j] = 0.0f; }
        stage_w(wl, wih, 1, H, g * H, 0);
        __syncthreads();
        mma_k128(Am, wl, ai);
        __syncthreads();
        stage_w(wl, whh, 1, H, g * H, 0);
        __syncthreads();
        mma_k128(Ah, wl, ah);
        __syncthreads();
#pragma unroll
        for (int t = 0; t < 8; ++t) {
            if (g < 2) grz[g][t] = ai[t] + ah[t];
            else       { gni[t] = ai[t]; gnh[t] = ah[t]; }
        }
    }
    int rsel = (lane & 16) ? 8 : 0;
#pragma unroll
    for (int t = 0; t < 8; ++t) {
        int col = t * 16 + (lane & 15);
        float b_r = bih[col] + bhh[col];
        float b_z = bih[H + col] + bhh[H + col];
        float b_i = bih[2 * H + col];
        float b_h = bhh[2 * H + col];
#pragma unroll
        for (int rv = 0; rv < 8; ++rv) {
            int row = r0 + rv + rsel;
            if (row < rows) {
                size_t o = (size_t)row * H + col;
                float hv = node[o];
                float rg = sigm_(grz[0][t][rv] + b_r);
                float zg = sigm_(grz[1][t][rv] + b_z);
                float ng = tanhf(gni[t][rv] + b_i + rg * (gnh[t][rv] + b_h));
                node[o] = (1.0f - zg) * ng + zg * hv;
            }
        }
    }
}

// Fused LSTM cell for Set2Set: qs[G,2H] in, hl/cl[G,H] updated in place.
__global__ __launch_bounds__(256) void k_lstm(
    const float* __restrict__ qs, float* __restrict__ hl, float* __restrict__ cl,
    const float* __restrict__ wih, const float* __restrict__ whh,
    const float* __restrict__ bih, const float* __restrict__ bhh, int rows) {
    __shared__ _Float16 wl[H * WSTRIDE];
    int wave = threadIdx.x >> 5, lane = threadIdx.x & 31;
    int r0 = blockIdx.x * 128 + wave * 16;
    int rowA = r0 + (lane & 15); if (rowA > rows - 1) rowA = rows - 1;
    const float* Aq = qs + (size_t)rowA * (2 * H);
    const float* Ah = hl + (size_t)rowA * H;

    v8f gacc[4][8];
#pragma unroll
    for (int g = 0; g < 4; ++g) {
#pragma unroll
        for (int t = 0; t < 8; ++t)
#pragma unroll
            for (int j = 0; j < 8; ++j) gacc[g][t][j] = 0.0f;
#pragma unroll
        for (int kc = 0; kc < 2; ++kc) {
            stage_w(wl, wih, 1, 2 * H, g * H, kc * 128);
            __syncthreads();
            mma_k128(Aq + kc * 128, wl, gacc[g]);
            __syncthreads();
        }
        stage_w(wl, whh, 1, H, g * H, 0);
        __syncthreads();
        mma_k128(Ah, wl, gacc[g]);
        __syncthreads();
    }
    int rsel = (lane & 16) ? 8 : 0;
#pragma unroll
    for (int t = 0; t < 8; ++t) {
        int col = t * 16 + (lane & 15);
        float b_i = bih[col] + bhh[col];
        float b_f = bih[H + col] + bhh[H + col];
        float b_g = bih[2 * H + col] + bhh[2 * H + col];
        float b_o = bih[3 * H + col] + bhh[3 * H + col];
#pragma unroll
        for (int rv = 0; rv < 8; ++rv) {
            int row = r0 + rv + rsel;
            if (row < rows) {
                size_t o = (size_t)row * H + col;
                float iv = sigm_(gacc[0][t][rv] + b_i);
                float fv = sigm_(gacc[1][t][rv] + b_f);
                float gv = tanhf(gacc[2][t][rv] + b_g);
                float ov = sigm_(gacc[3][t][rv] + b_o);
                float c  = fv * cl[o] + iv * gv;
                cl[o] = c;
                hl[o] = ov * tanhf(c);
            }
        }
    }
}

// out[n,c] = sum_k X[n,k]*W[k,c] + B[c]   (small-K input embeddings)
__global__ void k_embed(const float* __restrict__ X, int Kin,
                        const float* __restrict__ W, const float* __restrict__ B,
                        float* __restrict__ out, int rows) {
    int i = blockIdx.x * blockDim.x + threadIdx.x;
    if (i >= rows * H) return;
    int n = i >> 7, c = i & (H - 1);
    float s = B[c];
    for (int k = 0; k < Kin; ++k) s += X[(size_t)n * Kin + k] * W[k * H + c];
    out[i] = s;
}

// msg = relu(node[src] + edge); atomic scatter-add into agg[dst]
__global__ void k_scatter(const float* __restrict__ node, const float* __restrict__ edge,
                          const int* __restrict__ src, const int* __restrict__ dst,
                          float* __restrict__ agg, int E) {
    int i = blockIdx.x * blockDim.x + threadIdx.x;
    if (i >= E * 32) return;
    int e = i >> 5, c4 = (i & 31) << 2;
    int s = src[e], d = dst[e];
    float4 nv = *(const float4*)(node + (size_t)s * H + c4);
    float4 ev = *(const float4*)(edge + (size_t)e * H + c4);
    float* ap = agg + (size_t)d * H + c4;
    unsafeAtomicAdd(ap + 0, fmaxf(nv.x + ev.x, 0.0f));
    unsafeAtomicAdd(ap + 1, fmaxf(nv.y + ev.y, 0.0f));
    unsafeAtomicAdd(ap + 2, fmaxf(nv.z + ev.z, 0.0f));
    unsafeAtomicAdd(ap + 3, fmaxf(nv.w + ev.w, 0.0f));
}

__global__ void k_zcomb(const float* __restrict__ node, float* __restrict__ agg,
                        const float* __restrict__ eps, int l, int total) {
    int i = blockIdx.x * blockDim.x + threadIdx.x;
    if (i < total) agg[i] = (1.0f + eps[l]) * node[i] + agg[i];
}

// e[n] = dot(node[n,:], q[batch[n],:])  (one wave per node)
__global__ void k_attn_dot(const float* __restrict__ node, const float* __restrict__ q,
                           const int* __restrict__ batch, float* __restrict__ e, int N) {
    int wave = threadIdx.x >> 5, lane = threadIdx.x & 31;
    int n = blockIdx.x * 8 + wave;
    if (n >= N) return;
    int g = batch[n];
    float4 a = *(const float4*)(node + (size_t)n * H + lane * 4);
    float4 b = *(const float4*)(q + (size_t)g * H + lane * 4);
    float s = a.x * b.x + a.y * b.y + a.z * b.z + a.w * b.w;
#pragma unroll
    for (int o = 16; o > 0; o >>= 1) s += __shfl_xor(s, o, 32);
    if (lane == 0) e[n] = s;
}

__global__ void k_segmax(const float* __restrict__ e, const int* __restrict__ batch,
                         unsigned* __restrict__ emax, int N) {
    int i = blockIdx.x * blockDim.x + threadIdx.x;
    if (i < N) atomicMax(emax + batch[i], enc_ord(e[i]));
}

__global__ void k_attnexp(const float* __restrict__ e, const int* __restrict__ batch,
                          const unsigned* __restrict__ emax, float* __restrict__ exb,
                          float* __restrict__ denom, int N) {
    int i = blockIdx.x * blockDim.x + threadIdx.x;
    if (i >= N) return;
    int g = batch[i];
    float ex = expf(e[i] - dec_ord(emax[g]));
    exb[i] = ex;
    unsafeAtomicAdd(denom + g, ex);
}

__global__ void k_attnr(const float* __restrict__ node, const float* __restrict__ exb,
                        const float* __restrict__ denom, const int* __restrict__ batch,
                        float* __restrict__ rbuf, int N) {
    int i = blockIdx.x * blockDim.x + threadIdx.x;
    if (i >= N * 32) return;
    int n = i >> 5, c4 = (i & 31) << 2;
    int g = batch[n];
    float a = exb[n] / fmaxf(denom[g], 1e-9f);
    float4 nv = *(const float4*)(node + (size_t)n * H + c4);
    float* rp = rbuf + (size_t)g * H + c4;
    unsafeAtomicAdd(rp + 0, a * nv.x);
    unsafeAtomicAdd(rp + 1, a * nv.y);
    unsafeAtomicAdd(rp + 2, a * nv.z);
    unsafeAtomicAdd(rp + 3, a * nv.w);
}

__global__ void k_buildqs(const float* __restrict__ hl, const float* __restrict__ rbuf,
                          float* __restrict__ qs, int total) {
    int i = blockIdx.x * blockDim.x + threadIdx.x;
    if (i >= total) return;
    int g = i >> 7, c = i & (H - 1);
    qs[(size_t)g * 256 + c] = hl[i];
    qs[(size_t)g * 256 + H + c] = rbuf[i];
}

// out[g] = tfc[g,:] . w2 + b2   (one wave per graph)
__global__ void k_fc2(const float* __restrict__ tfc, const float* __restrict__ w2,
                      const float* __restrict__ b2, float* __restrict__ out, int G) {
    int wave = threadIdx.x >> 5, lane = threadIdx.x & 31;
    int g = blockIdx.x * 8 + wave;
    if (g >= G) return;
    float4 a = *(const float4*)(tfc + (size_t)g * H + lane * 4);
    float4 b = *(const float4*)(w2 + lane * 4);
    float s = a.x * b.x + a.y * b.y + a.z * b.z + a.w * b.w;
#pragma unroll
    for (int o = 16; o > 0; o >>= 1) s += __shfl_xor(s, o, 32);
    if (lane == 0) out[g] = s + b2[0];
}

static inline int cdiv(int a, int b) { return (a + b - 1) / b; }

extern "C" void kernel_launch(void* const* d_in, const int* in_sizes, int n_in,
                              void* d_out, int out_size, void* d_ws, size_t ws_size,
                              hipStream_t stream) {
    const float* x         = (const float*)d_in[0];
    const float* edge_attr = (const float*)d_in[1];
    const int*   eidx      = (const int*)d_in[2];
    const int*   batch     = (const int*)d_in[3];
    const float* node_w    = (const float*)d_in[4];
    const float* node_b    = (const float*)d_in[5];
    const float* edge_w    = (const float*)d_in[6];
    const float* edge_b    = (const float*)d_in[7];
    const float* eps       = (const float*)d_in[8];
    const float* gin_w1    = (const float*)d_in[9];
    const float* gin_b1    = (const float*)d_in[10];
    const float* gin_w2    = (const float*)d_in[11];
    const float* gin_b2    = (const float*)d_in[12];
    const float* em_w1     = (const float*)d_in[13];
    const float* em_b1     = (const float*)d_in[14];
    const float* em_w2     = (const float*)d_in[15];
    const float* em_b2     = (const float*)d_in[16];
    const float* gru_wih   = (const float*)d_in[17];
    const float* gru_whh   = (const float*)d_in[18];
    const float* gru_bih   = (const float*)d_in[19];
    const float* gru_bhh   = (const float*)d_in[20];
    const float* lstm_wih  = (const float*)d_in[21];
    const float* lstm_whh  = (const float*)d_in[22];
    const float* lstm_bih  = (const float*)d_in[23];
    const float* lstm_bhh  = (const float*)d_in[24];
    const float* fc_w1     = (const float*)d_in[25];
    const float* fc_b1     = (const float*)d_in[26];
    const float* fc_w2     = (const float*)d_in[27];
    const float* fc_b2     = (const float*)d_in[28];

    const int N = in_sizes[3];
    const int E = in_sizes[1] / 4;
    const int G = out_size;
    const int* src = eidx;
    const int* dst = eidx + E;

    float* ws = (float*)d_ws;
    float* node  = ws; ws += (size_t)N * H;
    float* edge  = ws; ws += (size_t)E * H;
    float* edge2 = ws; ws += (size_t)E * H;
    float* agg   = ws; ws += (size_t)N * H;
    float* t1    = ws; ws += (size_t)N * H;
    float* mbuf  = ws; ws += (size_t)N * H;
    float* ebuf  = ws; ws += N;
    float* exb   = ws; ws += N;
    float* qs    = ws; ws += (size_t)G * 2 * H;
    float* hl    = ws; ws += (size_t)G * H;
    float* cl    = ws; ws += (size_t)G * H;
    float* rbuf  = ws; ws += (size_t)G * H;
    float* tfc   = ws; ws += (size_t)G * H;
    float* denom = ws; ws += G;
    unsigned* emax = (unsigned*)ws;

    // input embeddings
    k_embed<<<cdiv(N * H, 256), 256, 0, stream>>>(x, 14, node_w, node_b, node, N);
    k_embed<<<cdiv(E * H, 256), 256, 0, stream>>>(edge_attr, 4, edge_w, edge_b, edge, E);

    const int L = 2;
    for (int l = 0; l < L; ++l) {
        hipMemsetAsync(agg, 0, (size_t)N * H * sizeof(float), stream);
        k_scatter<<<cdiv(E * 32, 256), 256, 0, stream>>>(node, edge, src, dst, agg, E);
        k_zcomb<<<cdiv(N * H, 256), 256, 0, stream>>>(node, agg, eps, l, N * H);
        k_wmma_gemm<<<cdiv(N, 128), 256, 0, stream>>>(agg, N, H, gin_w1 + (size_t)l * H * H, 0, H,
                                                      gin_b1 + l * H, t1, 1);
        k_wmma_gemm<<<cdiv(N, 128), 256, 0, stream>>>(t1, N, H, gin_w2 + (size_t)l * H * H, 0, H,
                                                      gin_b2 + l * H, mbuf, 0);
        k_gru<<<cdiv(N, 128), 256, 0, stream>>>(node, mbuf, gru_wih, gru_whh, gru_bih, gru_bhh, N);
        k_wmma_gemm<<<cdiv(E, 128), 256, 0, stream>>>(edge, E, H, em_w1 + (size_t)l * H * H, 0, H,
                                                      em_b1 + l * H, edge2, 1);
        k_wmma_gemm<<<cdiv(E, 128), 256, 0, stream>>>(edge2, E, H, em_w2 + (size_t)l * H * H, 0, H,
                                                      em_b2 + l * H, edge, 0);
    }

    // Set2Set pooling, 3 steps
    hipMemsetAsync(qs, 0, (size_t)G * 2 * H * sizeof(float), stream);
    hipMemsetAsync(hl, 0, (size_t)G * H * sizeof(float), stream);
    hipMemsetAsync(cl, 0, (size_t)G * H * sizeof(float), stream);
    for (int s3 = 0; s3 < 3; ++s3) {
        k_lstm<<<cdiv(G, 128), 256, 0, stream>>>(qs, hl, cl, lstm_wih, lstm_whh,
                                                 lstm_bih, lstm_bhh, G);
        k_attn_dot<<<cdiv(N, 8), 256, 0, stream>>>(node, hl, batch, ebuf, N);
        hipMemsetAsync(emax, 0, (size_t)G * sizeof(unsigned), stream);
        hipMemsetAsync(denom, 0, (size_t)G * sizeof(float), stream);
        hipMemsetAsync(rbuf, 0, (size_t)G * H * sizeof(float), stream);
        k_segmax<<<cdiv(N, 256), 256, 0, stream>>>(ebuf, batch, emax, N);
        k_attnexp<<<cdiv(N, 256), 256, 0, stream>>>(ebuf, batch, emax, exb, denom, N);
        k_attnr<<<cdiv(N * 32, 256), 256, 0, stream>>>(node, exb, denom, batch, rbuf, N);
        k_buildqs<<<cdiv(G * H, 256), 256, 0, stream>>>(hl, rbuf, qs, G * H);
    }

    // final MLP: relu(qs @ fc_w1 + b1) @ fc_w2 + b2
    k_wmma_gemm<<<cdiv(G, 128), 256, 0, stream>>>(qs, G, 2 * H, fc_w1, 0, H, fc_b1, tfc, 1);
    k_fc2<<<cdiv(G, 8), 256, 0, stream>>>(tfc, fc_w2, fc_b2, (float*)d_out, G);
}
